// TraceablePhimoeSparseMoeBlock_86328842649607
// MI455X (gfx1250) — matrix operand order repaired and checked by reading, not compiled, not run on hardware
//
#include <hip/hip_runtime.h>

// ---------------- Problem constants (from reference) ----------------
static constexpr int N    = 2 * 1024;   // B*S tokens
static constexpr int H    = 1024;       // hidden
static constexpr int F    = 2816;       // ffn
static constexpr int E    = 8;          // experts
static constexpr int TK   = 2;          // top-k
static constexpr int CAP  = N * TK + E * 16;   // padded assignment slots = 4224
static constexpr int MAXT = CAP / 16;          // max M-tiles = 264
static constexpr int WELEM = E * F * H;        // elements per weight tensor = 23,068,672

// ---------------- WMMA vector types (gfx1250, wave32) ----------------
typedef __attribute__((ext_vector_type(16))) __bf16 v16bf;
typedef __attribute__((ext_vector_type(8)))  __bf16 v8bf;
typedef __attribute__((ext_vector_type(8)))  float  v8f;
typedef __attribute__((ext_vector_type(4)))  float  v4f;

// ---------------- Workspace layout (bytes, 256-aligned) ----------------
static constexpr size_t ALGN = 256;
static constexpr size_t rup(size_t x) { return (x + ALGN - 1) & ~(ALGN - 1); }
static constexpr size_t OFF_TOPI = 0;
static constexpr size_t OFF_TOPW = OFF_TOPI + rup((size_t)N * TK * 4);
static constexpr size_t OFF_CNT  = OFF_TOPW + rup((size_t)N * TK * 4);
static constexpr size_t OFF_FILL = OFF_CNT  + rup(E * 4);
static constexpr size_t OFF_NUMT = OFF_FILL + rup(E * 4);
static constexpr size_t OFF_TILE = OFF_NUMT + rup(4);
static constexpr size_t OFF_TOK  = OFF_TILE + rup((size_t)MAXT * 2 * 4);
static constexpr size_t OFF_KK   = OFF_TOK  + rup((size_t)CAP * 4);
static constexpr size_t OFF_SLW  = OFF_KK   + rup((size_t)CAP * 4);
static constexpr size_t OFF_XG   = OFF_SLW  + rup((size_t)CAP * 4);
static constexpr size_t OFF_ACT  = OFF_XG   + rup((size_t)CAP * H * 2);
static constexpr size_t OFF_PART = OFF_ACT  + rup((size_t)CAP * F * 2);
static constexpr size_t OFF_W1B  = OFF_PART + rup((size_t)N * TK * H * 4);
static constexpr size_t OFF_W2B  = OFF_W1B  + rup((size_t)WELEM * 2);
// total ~141 MB

// ===================================================================
// 0) init: zero expert counts, invalidate all gather slots
// ===================================================================
__global__ void moe_init_kernel(int* __restrict__ cnt, int* __restrict__ tok_slot) {
    int j = blockIdx.x * blockDim.x + threadIdx.x;
    if (j < CAP) tok_slot[j] = -1;
    if (j < E)   cnt[j] = 0;
}

// ===================================================================
// 0b) one-time f32 -> bf16 weight conversion (L2-resident afterwards)
// ===================================================================
__global__ __launch_bounds__(256)
void cvt_bf16_kernel(const float* __restrict__ src, __bf16* __restrict__ dst, int n8) {
    int i = blockIdx.x * 256 + threadIdx.x;
    if (i >= n8) return;
    v4f s0 = *(const v4f*)(src + (size_t)i * 8);
    v4f s1 = *(const v4f*)(src + (size_t)i * 8 + 4);
    v8bf d;
#pragma unroll
    for (int j = 0; j < 4; ++j) { d[j] = (__bf16)s0[j]; d[4 + j] = (__bf16)s1[j]; }
    *(v8bf*)(dst + (size_t)i * 8) = d;
}

// ===================================================================
// 1) router: one wave per token. logits -> softmax -> top2 -> renorm
// ===================================================================
__global__ __launch_bounds__(256)
void moe_router_kernel(const float* __restrict__ x, const float* __restrict__ gw,
                       int* __restrict__ topi, float* __restrict__ topw,
                       int* __restrict__ cnt) {
    const int lane = threadIdx.x & 31;
    const int n    = blockIdx.x * 8 + (threadIdx.x >> 5);
    if (n >= N) return;

    float acc[E];
#pragma unroll
    for (int e = 0; e < E; ++e) acc[e] = 0.f;

    const float* xr = x + (size_t)n * H;
    for (int h = lane; h < H; h += 32) {
        float xv = xr[h];
#pragma unroll
        for (int e = 0; e < E; ++e) acc[e] += xv * gw[e * H + h];
    }
#pragma unroll
    for (int off = 16; off >= 1; off >>= 1) {
#pragma unroll
        for (int e = 0; e < E; ++e) acc[e] += __shfl_xor(acc[e], off, 32);
    }
    if (lane == 0) {
        float mx = acc[0];
#pragma unroll
        for (int e = 1; e < E; ++e) mx = fmaxf(mx, acc[e]);
        float p[E];
#pragma unroll
        for (int e = 0; e < E; ++e) p[e] = __expf(acc[e] - mx);
        int i0 = 0;
#pragma unroll
        for (int e = 1; e < E; ++e) if (p[e] > p[i0]) i0 = e;
        int i1 = (i0 == 0) ? 1 : 0;
#pragma unroll
        for (int e = 0; e < E; ++e) if (e != i0 && p[e] > p[i1]) i1 = e;
        float s = p[i0] + p[i1];
        topi[n * TK + 0] = i0;  topw[n * TK + 0] = p[i0] / s;
        topi[n * TK + 1] = i1;  topw[n * TK + 1] = p[i1] / s;
        atomicAdd(&cnt[i0], 1);
        atomicAdd(&cnt[i1], 1);
    }
}

// ===================================================================
// 2) prefix: 16-padded expert bases + flat tile descriptor table
// ===================================================================
__global__ void moe_prefix_kernel(const int* __restrict__ cnt, int* __restrict__ fill,
                                  int* __restrict__ tiledesc, int* __restrict__ numTiles) {
    if (threadIdx.x != 0 || blockIdx.x != 0) return;
    int b = 0, t = 0;
    for (int e = 0; e < E; ++e) {
        fill[e] = b;
        int mt = (cnt[e] + 15) >> 4;
        for (int i = 0; i < mt; ++i) { tiledesc[2 * t] = e; tiledesc[2 * t + 1] = b + 16 * i; ++t; }
        b += mt * 16;
    }
    numTiles[0] = t;
}

// ===================================================================
// 3) scatter: fill per-expert slot lists
// ===================================================================
__global__ void moe_scatter_kernel(const int* __restrict__ topi, const float* __restrict__ topw,
                                   int* __restrict__ fill, int* __restrict__ tok_slot,
                                   int* __restrict__ kk_slot, float* __restrict__ slot_w) {
    int n = blockIdx.x * blockDim.x + threadIdx.x;
    if (n >= N) return;
#pragma unroll
    for (int k = 0; k < TK; ++k) {
        int e = topi[n * TK + k];
        int j = atomicAdd(&fill[e], 1);
        tok_slot[j] = n;
        kk_slot[j]  = k;
        slot_w[j]   = topw[n * TK + k];
    }
}

// ===================================================================
// 4) gather: bf16-convert gathered token rows; zero padded rows
// ===================================================================
__global__ __launch_bounds__(256)
void moe_gather_kernel(const float* __restrict__ x, const int* __restrict__ tok_slot,
                       __bf16* __restrict__ xg) {
    int slot = blockIdx.x;
    int tok  = tok_slot[slot];
    __bf16* dst = xg + (size_t)slot * H;
    if (tok >= 0) {
        const float* src = x + (size_t)tok * H;
        for (int h = threadIdx.x; h < H; h += 256) dst[h] = (__bf16)src[h];
    } else {
        for (int h = threadIdx.x; h < H; h += 256) dst[h] = (__bf16)0.f;
    }
}

// ---------------- fragment helpers (ISA 16-bit A 16x32 layout) ----------------
// lane L: row/col = L&15, half = L>>4.
// A: K elems = [kk + half*8 .. +7] then [kk + 16 + half*8 .. +7]   (arow includes +half*8)
// B: K elems = [kk + half*8 .. +7] then [kk + 16 + half*8 .. +7]   (brow includes +half*8)
__device__ inline v16bf pack_a(const __bf16* arow, int kk) {
    v8bf a0 = *(const v8bf*)(arow + kk);
    v8bf a1 = *(const v8bf*)(arow + kk + 16);
    v16bf a;
#pragma unroll
    for (int i = 0; i < 8; ++i) { a[i] = a0[i]; a[8 + i] = a1[i]; }
    return a;
}
__device__ inline v16bf pack_b(const __bf16* brow, int kk) {
    v8bf b0 = *(const v8bf*)(brow + kk);
    v8bf b1 = *(const v8bf*)(brow + kk + 16);
    v16bf b;
#pragma unroll
    for (int i = 0; i < 8; ++i) { b[i] = b0[i]; b[8 + i] = b1[i]; }
    return b;
}

// ===================================================================
// 5) up-proj + SiLU: act[slot, f] = silu(xg[slot,:] @ w1[e][f,:])
//    grid: (MAXT, F/256), block 128. Each wave: 16x64 strip (4 N-tiles,
//    one shared A fragment -> 4 WMMA per K-step).
// ===================================================================
__global__ __launch_bounds__(128)
void moe_up_kernel(const __bf16* __restrict__ xg, const __bf16* __restrict__ w1b,
                   const int* __restrict__ tiledesc, const int* __restrict__ numTiles,
                   __bf16* __restrict__ act) {
    const int t = blockIdx.x;
    if (t >= numTiles[0]) return;
    const int wave = threadIdx.x >> 5, lane = threadIdx.x & 31;
    const int e = tiledesc[2 * t], row0 = tiledesc[2 * t + 1];
    const int fbase = blockIdx.y * 256 + wave * 64;
    const int mn = lane & 15, hl = lane >> 4;

    const __bf16* arow = xg + (size_t)(row0 + mn) * H + hl * 8;
    const __bf16* b0   = w1b + ((size_t)e * F + (fbase + mn)) * H + hl * 8;

    v8f c0 = {}, c1 = {}, c2 = {}, c3 = {};
#pragma unroll 2
    for (int kk = 0; kk < H; kk += 32) {
        v16bf a = pack_a(arow, kk);
        c0 = __builtin_amdgcn_wmma_f32_16x16x32_bf16(false, a, false,
                 pack_b(b0 + (size_t)0 * 16 * H, kk), (short)0, c0, false, false);
        c1 = __builtin_amdgcn_wmma_f32_16x16x32_bf16(false, a, false,
                 pack_b(b0 + (size_t)1 * 16 * H, kk), (short)0, c1, false, false);
        c2 = __builtin_amdgcn_wmma_f32_16x16x32_bf16(false, a, false,
                 pack_b(b0 + (size_t)2 * 16 * H, kk), (short)0, c2, false, false);
        c3 = __builtin_amdgcn_wmma_f32_16x16x32_bf16(false, a, false,
                 pack_b(b0 + (size_t)3 * 16 * H, kk), (short)0, c3, false, false);
    }
    // SiLU + store (D layout: VGPR r -> row M = r + 8*half, col N = lane&15)
    v8f cc[4] = {c0, c1, c2, c3};
#pragma unroll
    for (int r = 0; r < 8; ++r) {
        int slot = row0 + r + hl * 8;
        __bf16* drow = act + (size_t)slot * F + fbase + mn;
#pragma unroll
        for (int j = 0; j < 4; ++j) {
            float v  = cc[j][r];
            float sv = v * (1.f / (1.f + __expf(-v)));
            drow[j * 16] = (__bf16)sv;
        }
    }
}

// ===================================================================
// 6) down-proj + weight: partial[tok,k,h] = w * (act[slot,:] @ w2[e][h,:])
//    grid: (MAXT, H/256), block 128. 16x64 strip per wave.
// ===================================================================
__global__ __launch_bounds__(128)
void moe_down_kernel(const __bf16* __restrict__ act, const __bf16* __restrict__ w2b,
                     const int* __restrict__ tiledesc, const int* __restrict__ numTiles,
                     const int* __restrict__ tok_slot, const int* __restrict__ kk_slot,
                     const float* __restrict__ slot_w, float* __restrict__ partial) {
    const int t = blockIdx.x;
    if (t >= numTiles[0]) return;
    const int wave = threadIdx.x >> 5, lane = threadIdx.x & 31;
    const int e = tiledesc[2 * t], row0 = tiledesc[2 * t + 1];
    const int hbase = blockIdx.y * 256 + wave * 64;
    const int mn = lane & 15, hl = lane >> 4;

    const __bf16* arow = act + (size_t)(row0 + mn) * F + hl * 8;
    const __bf16* b0   = w2b + ((size_t)e * H + (hbase + mn)) * F + hl * 8;

    v8f c0 = {}, c1 = {}, c2 = {}, c3 = {};
#pragma unroll 2
    for (int kk = 0; kk < F; kk += 32) {   // 88 K-steps
        v16bf a = pack_a(arow, kk);
        c0 = __builtin_amdgcn_wmma_f32_16x16x32_bf16(false, a, false,
                 pack_b(b0 + (size_t)0 * 16 * F, kk), (short)0, c0, false, false);
        c1 = __builtin_amdgcn_wmma_f32_16x16x32_bf16(false, a, false,
                 pack_b(b0 + (size_t)1 * 16 * F, kk), (short)0, c1, false, false);
        c2 = __builtin_amdgcn_wmma_f32_16x16x32_bf16(false, a, false,
                 pack_b(b0 + (size_t)2 * 16 * F, kk), (short)0, c2, false, false);
        c3 = __builtin_amdgcn_wmma_f32_16x16x32_bf16(false, a, false,
                 pack_b(b0 + (size_t)3 * 16 * F, kk), (short)0, c3, false, false);
    }
    v8f cc[4] = {c0, c1, c2, c3};
#pragma unroll
    for (int r = 0; r < 8; ++r) {
        int slot = row0 + r + hl * 8;
        int tok  = tok_slot[slot];
        if (tok < 0) continue;
        float w = slot_w[slot];
        float* drow = partial + ((size_t)tok * TK + kk_slot[slot]) * H + hbase + mn;
#pragma unroll
        for (int j = 0; j < 4; ++j) drow[j * 16] = cc[j][r] * w;
    }
}

// ===================================================================
// 7) combine: y[n,h] = partial[n,0,h] + partial[n,1,h]
// ===================================================================
__global__ void moe_combine_kernel(const float* __restrict__ partial, float* __restrict__ out) {
    int idx = blockIdx.x * blockDim.x + threadIdx.x;
    if (idx >= N * H) return;
    int n = idx >> 10;          // /H
    int h = idx & (H - 1);
    out[idx] = partial[((size_t)n * TK + 0) * H + h] + partial[((size_t)n * TK + 1) * H + h];
}

// ===================================================================
extern "C" void kernel_launch(void* const* d_in, const int* in_sizes, int n_in,
                              void* d_out, int out_size, void* d_ws, size_t ws_size,
                              hipStream_t stream) {
    const float* x  = (const float*)d_in[0];   // [N,H]
    const float* gw = (const float*)d_in[1];   // [E,H]
    const float* w1 = (const float*)d_in[2];   // [E,F,H]
    const float* w2 = (const float*)d_in[3];   // [E,H,F]
    float* out = (float*)d_out;

    char* ws = (char*)d_ws;
    int*    topi     = (int*)(ws + OFF_TOPI);
    float*  topw     = (float*)(ws + OFF_TOPW);
    int*    cnt      = (int*)(ws + OFF_CNT);
    int*    fill     = (int*)(ws + OFF_FILL);
    int*    numTiles = (int*)(ws + OFF_NUMT);
    int*    tiledesc = (int*)(ws + OFF_TILE);
    int*    tok_slot = (int*)(ws + OFF_TOK);
    int*    kk_slot  = (int*)(ws + OFF_KK);
    float*  slot_w   = (float*)(ws + OFF_SLW);
    __bf16* xg       = (__bf16*)(ws + OFF_XG);
    __bf16* act      = (__bf16*)(ws + OFF_ACT);
    float*  partial  = (float*)(ws + OFF_PART);
    __bf16* w1b      = (__bf16*)(ws + OFF_W1B);
    __bf16* w2b      = (__bf16*)(ws + OFF_W2B);

    constexpr int N8 = WELEM / 8;  // 8 f32 per conversion thread

    moe_init_kernel<<<(CAP + 255) / 256, 256, 0, stream>>>(cnt, tok_slot);
    cvt_bf16_kernel<<<(N8 + 255) / 256, 256, 0, stream>>>(w1, w1b, N8);
    cvt_bf16_kernel<<<(N8 + 255) / 256, 256, 0, stream>>>(w2, w2b, N8);
    moe_router_kernel<<<N / 8, 256, 0, stream>>>(x, gw, topi, topw, cnt);
    moe_prefix_kernel<<<1, 1, 0, stream>>>(cnt, fill, tiledesc, numTiles);
    moe_scatter_kernel<<<(N + 255) / 256, 256, 0, stream>>>(topi, topw, fill, tok_slot,
                                                            kk_slot, slot_w);
    moe_gather_kernel<<<CAP, 256, 0, stream>>>(x, tok_slot, xg);
    moe_up_kernel<<<dim3(MAXT, F / 256), 128, 0, stream>>>(xg, w1b, tiledesc, numTiles, act);
    moe_down_kernel<<<dim3(MAXT, H / 256), 128, 0, stream>>>(act, w2b, tiledesc, numTiles,
                                                             tok_slot, kk_slot, slot_w, partial);
    moe_combine_kernel<<<(N * H + 255) / 256, 256, 0, stream>>>(partial, out);
}
